// HeterogeneousFraudGNNSageMaker_41386304864315
// MI455X (gfx1250) — matrix operand order repaired and checked by reading, not compiled, not run on hardware
//
#include <hip/hip_runtime.h>
#include <math.h>

// ---------------------------------------------------------------------------
// Types for CDNA5 WMMA (wave32): bf16 A/B fragments, f32 accumulator
// ---------------------------------------------------------------------------
typedef __attribute__((ext_vector_type(16))) __bf16 v16bf;
typedef __attribute__((ext_vector_type(8)))  __bf16 v8bf;
typedef __attribute__((ext_vector_type(4)))  __bf16 v4bf;
typedef __attribute__((ext_vector_type(8)))  float  v8f;

#define GF_BIAS 1
#define GF_RELU 2
#define GF_ACC  4

__device__ __forceinline__ __bf16 f2bf(float f) { return (__bf16)f; }

// two contiguous 16B runs in LDS -> one v16bf fragment (2x ds_load_b128)
__device__ __forceinline__ v16bf ld_frag(const __bf16* p0, const __bf16* p1) {
    v8bf a = *(const v8bf*)p0;
    v8bf b = *(const v8bf*)p1;
    return __builtin_shufflevector(a, b, 0, 1, 2, 3, 4, 5, 6, 7,
                                         8, 9, 10, 11, 12, 13, 14, 15);
}

// ---------------------------------------------------------------------------
// WMMA GEMM: C[N,NOa] (+= / =) A[N,lda(K)] * W[K,NOa] (+bias) (ReLU)
// Block = 256 threads = 8 waves; 256 rows per block, 32 rows (2 strips) per
// wave, NT 16-col tiles. A staged row-major bf16; W staged TRANSPOSED
// (sWt[col][k]) so B fragments are contiguous 16B runs.
// ---------------------------------------------------------------------------
template <int NT>
__global__ __launch_bounds__(256)
void gemm_wmma(const float* __restrict__ A, int lda,
               const float* __restrict__ W,
               const float* __restrict__ bias,
               float* __restrict__ C,
               int N, int K, int NOa, int flags)
{
    constexpr int NOp = NT * 16;
    constexpr int SAS = 40;                    // sA row stride (bf16), 80B, 16B-aligned runs
    __shared__ alignas(16) __bf16 sA[256][SAS];
    __shared__ alignas(16) __bf16 sWt[NOp][SAS];   // transposed: [col][k]

    const int tid  = threadIdx.x;
    const int wave = tid >> 5;
    const int lane = tid & 31;
    const int half = lane >> 4;                // lane group 0/1
    const int nl   = lane & 15;
    const int rowBase = blockIdx.x * 256;

    v8f acc0[NT], acc1[NT];
#pragma unroll
    for (int t = 0; t < NT; ++t)
#pragma unroll
        for (int r = 0; r < 8; ++r) { acc0[t][r] = 0.0f; acc1[t][r] = 0.0f; }

    const int Kp = (K + 31) & ~31;
    for (int k0 = 0; k0 < Kp; k0 += 32) {
        __syncthreads();
        // ---- stage A chunk: 256 rows x 32 k, quad-wide (float4 -> v4bf) ----
        for (int i = tid; i < 256 * 8; i += 256) {
            int r = i >> 3, q = i & 7;
            int row = rowBase + r, k = k0 + q * 4;
            v4bf o;
            if (row < N && k + 3 < K) {
                const float4 v = *(const float4*)(A + (size_t)row * lda + k);
                o[0] = f2bf(v.x); o[1] = f2bf(v.y); o[2] = f2bf(v.z); o[3] = f2bf(v.w);
            } else {
#pragma unroll
                for (int j = 0; j < 4; ++j)
                    o[j] = f2bf((row < N && k + j < K) ? A[(size_t)row * lda + k + j] : 0.0f);
            }
            *(v4bf*)&sA[r][q * 4] = o;
        }
        // ---- stage W chunk transposed: sWt[n][k], quad of k per store ----
        for (int i = tid; i < NOp * 8; i += 256) {
            int n = i >> 3, q = i & 7;
            int k = k0 + q * 4;
            v4bf o;
#pragma unroll
            for (int j = 0; j < 4; ++j) {
                int kk = k + j;
                o[j] = f2bf((kk < K && n < NOa) ? W[(size_t)kk * NOa + n] : 0.0f);
            }
            *(v4bf*)&sWt[n][q * 4] = o;
        }
        __syncthreads();

        // ---- A fragments for this wave's two 16-row strips ----
        // lane m = row%16; grp0 holds K {0..7,16..23}, grp1 holds K {8..15,24..31}
        const int m0 = (wave << 5) + nl;
        const int m1 = m0 + 16;
        const v16bf af0 = ld_frag(&sA[m0][half << 3], &sA[m0][16 + (half << 3)]);
        const v16bf af1 = ld_frag(&sA[m1][half << 3], &sA[m1][16 + (half << 3)]);

#pragma unroll
        for (int t = 0; t < NT; ++t) {
            // B fragment: col n = L%16; grp0: K {0..7,16..23}, grp1: K {8..15,24..31}
            const int n  = (t << 4) + nl;
            const int kb = half << 3;
            const v16bf bfv = ld_frag(&sWt[n][kb], &sWt[n][kb + 16]);
            acc0[t] = __builtin_amdgcn_wmma_f32_16x16x32_bf16(
                false, af0, false, bfv, (short)0, acc0[t], false, false);
            acc1[t] = __builtin_amdgcn_wmma_f32_16x16x32_bf16(
                false, af1, false, bfv, (short)0, acc1[t], false, false);
        }
    }

    // ---- epilogue: C/D layout — lane n=L%16, VGPR r -> row M = r + 8*(L/16) ----
#pragma unroll
    for (int s = 0; s < 2; ++s) {
#pragma unroll
        for (int t = 0; t < NT; ++t) {
            const int n = (t << 4) + nl;
            if (n >= NOa) continue;
            const float bv = (flags & GF_BIAS) ? bias[n] : 0.0f;
#pragma unroll
            for (int r = 0; r < 8; ++r) {
                const int row = rowBase + (wave << 5) + (s << 4) + (half << 3) + r;
                if (row < N) {
                    size_t idx = (size_t)row * NOa + n;
                    float v = (s ? acc1[t][r] : acc0[t][r]) + bv;
                    if (flags & GF_ACC)  v += C[idx];
                    if (flags & GF_RELU) v = fmaxf(v, 0.0f);
                    C[idx] = v;
                }
            }
        }
    }
}

// ---------------------------------------------------------------------------
// Elementwise / edge-parallel kernels
// ---------------------------------------------------------------------------
__global__ void fill_f32(float* p, float v, long long n) {
    long long i = (long long)blockIdx.x * 256 + threadIdx.x;
    if (i < n) p[i] = v;
}
__global__ void fill_u32(unsigned* p, unsigned v, long long n) {
    long long i = (long long)blockIdx.x * 256 + threadIdx.x;
    if (i < n) p[i] = v;
}
__global__ void seg_count(const int* __restrict__ sidx, int E, float* cnt) {
    long long i = (long long)blockIdx.x * 256 + threadIdx.x;
    if (i < E) atomicAdd(cnt + sidx[i], 1.0f);
}
__global__ void seg_scatter_sum(const float* __restrict__ feat,
                                const int* __restrict__ gidx,
                                const int* __restrict__ sidx,
                                int E, float* agg) {
    long long i = (long long)blockIdx.x * 256 + threadIdx.x;
    if (i >= (long long)E * 128) return;
    int c = (int)(i & 127); int e = (int)(i >> 7);
    atomicAdd(agg + (size_t)sidx[e] * 128 + c, feat[(size_t)gidx[e] * 128 + c]);
}
__global__ void seg_normalize(float* agg, const float* cnt, long long n128) {
    long long i = (long long)blockIdx.x * 256 + threadIdx.x;
    if (i < n128) agg[i] /= fmaxf(cnt[i >> 7], 1.0f);
}
__global__ void bn_relu_k(float* x, const float* g, const float* b, long long n128) {
    long long i = (long long)blockIdx.x * 256 + threadIdx.x;
    if (i >= n128) return;
    int c = (int)(i & 127);
    float rs = rsqrtf(1.0f + 1e-5f);
    x[i] = fmaxf(x[i] * (g[c] * rs) + b[c], 0.0f);
}
__global__ void gat_bn_relu_k(float* x, const float* bg, const float* g,
                              const float* b, long long n128) {
    long long i = (long long)blockIdx.x * 256 + threadIdx.x;
    if (i >= n128) return;
    int c = (int)(i & 127);
    float rs = rsqrtf(1.0f + 1e-5f);
    float v = x[i] + bg[c];
    x[i] = fmaxf(v * (g[c] * rs) + b[c], 0.0f);
}
__global__ void gat_scores_k(const float* __restrict__ hs,
                             const float* __restrict__ attS,
                             const float* __restrict__ attD,
                             float* as_, float* ad_, int N) {
    long long i = (long long)blockIdx.x * 256 + threadIdx.x;
    if (i >= (long long)N * 4) return;
    int h = (int)(i & 3); int n = (int)(i >> 2);
    const float* row = hs + (size_t)n * 128 + h * 32;
    float s = 0.0f, d = 0.0f;
    for (int j = 0; j < 32; ++j) {
        float v = row[j];
        s += v * attS[h * 32 + j];
        d += v * attD[h * 32 + j];
    }
    as_[i] = s; ad_[i] = d;
}
__device__ __forceinline__ unsigned fflip(float f) {
    unsigned u = __float_as_uint(f);
    return (u & 0x80000000u) ? ~u : (u | 0x80000000u);
}
__device__ __forceinline__ float funflip(unsigned u) {
    return (u & 0x80000000u) ? __uint_as_float(u & 0x7fffffffu) : __uint_as_float(~u);
}
__device__ __forceinline__ float leaky02(float v) { return v >= 0.0f ? v : 0.2f * v; }

__global__ void gat_edge_max_k(const float* as_, const float* ad_,
                               const int* __restrict__ ei, int E, unsigned* mb) {
    long long i = (long long)blockIdx.x * 256 + threadIdx.x;
    if (i >= (long long)E * 4) return;
    int h = (int)(i & 3); int e = (int)(i >> 2);
    int s = ei[e], d = ei[E + e];
    float v = leaky02(as_[(size_t)s * 4 + h] + ad_[(size_t)d * 4 + h]);
    atomicMax(mb + (size_t)d * 4 + h, fflip(v));
}
__global__ void gat_m_fin_k(const unsigned* mbu, float* mbf, float* den, long long n4) {
    long long i = (long long)blockIdx.x * 256 + threadIdx.x;
    if (i >= n4) return;
    float m = funflip(mbu[i]);
    if (m < -3.0e38f || m > 3.0e38f) m = 0.0f;   // isfinite replacement
    mbf[i] = m; den[i] = 0.0f;
}
__global__ void gat_edge_exp_k(const float* as_, const float* ad_,
                               const int* __restrict__ ei, int E,
                               const float* mbf, float* den, float* exb) {
    long long i = (long long)blockIdx.x * 256 + threadIdx.x;
    if (i >= (long long)E * 4) return;
    int h = (int)(i & 3); int e = (int)(i >> 2);
    int s = ei[e], d = ei[E + e];
    float v  = leaky02(as_[(size_t)s * 4 + h] + ad_[(size_t)d * 4 + h]);
    float ex = expf(v - mbf[(size_t)d * 4 + h]);
    exb[i] = ex;
    atomicAdd(den + (size_t)d * 4 + h, ex);
}
__global__ void gat_scatter_k(const float* __restrict__ hs,
                              const int* __restrict__ ei, int E,
                              const float* exb, const float* den, float* out) {
    long long i = (long long)blockIdx.x * 256 + threadIdx.x;
    if (i >= (long long)E * 128) return;
    int c = (int)(i & 127); int e = (int)(i >> 7);
    int s = ei[e], d = ei[E + e];
    int h = c >> 5;
    float alpha = exb[(size_t)e * 4 + h] / den[(size_t)d * 4 + h];
    atomicAdd(out + (size_t)d * 128 + c, alpha * hs[(size_t)s * 128 + c]);
}
__global__ void ctx_axpy_k(float* ctxw, const float* agg, const float* cnt,
                           const float* wsm, int slot, long long n128) {
    long long i = (long long)blockIdx.x * 256 + threadIdx.x;
    if (i >= n128) return;
    float cm = agg[i] / fmaxf(cnt[i >> 7], 1.0f);
    ctxw[i] += wsm[slot] * cm;
}
__global__ void softmax5_k(const float* a, float* w) {
    if (threadIdx.x == 0 && blockIdx.x == 0) {
        float m = a[0];
        for (int i = 1; i < 5; ++i) m = fmaxf(m, a[i]);
        float e[5], s = 0.0f;
        for (int i = 0; i < 5; ++i) { e[i] = expf(a[i] - m); s += e[i]; }
        for (int i = 0; i < 5; ++i) w[i] = e[i] / s;
    }
}
__global__ void logsoftmax3_k(const float* lg, float* out, int N) {
    long long i = (long long)blockIdx.x * 256 + threadIdx.x;
    if (i >= N) return;
    float a = lg[i * 3], b = lg[i * 3 + 1], c = lg[i * 3 + 2];
    float m = fmaxf(a, fmaxf(b, c));
    float l = logf(expf(a - m) + expf(b - m) + expf(c - m)) + m;
    out[i * 3] = a - l; out[i * 3 + 1] = b - l; out[i * 3 + 2] = c - l;
}

// ---------------------------------------------------------------------------
// Flat param offsets: jax tree-flatten order (dict keys sorted alphabetically)
// params = {cls, emb, info_att, layers}
// ---------------------------------------------------------------------------
static const size_t CLS_W1 = 0;                       // [256,128]
static const size_t CLS_W2 = 32768;                   // [128,64]
static const size_t CLS_W3 = 40960;                   // [64,3]
static const size_t CLS_B1 = 41152, CLS_B2 = 41280, CLS_B3 = 41344;
static const size_t EMB_COM_W = 41347, EMB_COM_B = 45443;   // comercializadora [32,128]
static const size_t EMB_CON_W = 45571, EMB_CON_B = 49667;   // concentrador     [32,128]
static const size_t EMB_CTA_W = 49795, EMB_CTA_B = 66179;   // contador        [128,128]
static const size_t EMB_SUM_W = 72579, EMB_SUM_B = 80771;   // suministro       [64,128]
static const size_t EMB_UBI_W = 80899, EMB_UBI_B = 82947;   // ubicacion        [16,128]
static const size_t INFO_ATT  = 83075;                      // [5]
static const size_t LBASE = 83080, LSTRIDE = 149632;
// per-layer relative offsets (keys sorted: bn, cerca, comunica, contratado, mide, ubicado)
static const size_t BN_COM_G=0,  BN_COM_B=128, BN_CON_G=256,  BN_CON_B=384;
static const size_t BN_CTA_G=512,BN_CTA_B=640, BN_SUM_G=768,  BN_SUM_B=896;
static const size_t BN_UBI_G=1024, BN_UBI_B=1152;
static const size_t GAT_W=1280, GAT_ATT_D=17664, GAT_ATT_S=17792, GAT_B=17920;
static const size_t COMU_WL=18048,  COMU_WR=34432,  COMU_BL=50816;
static const size_t CONT_WL=50944,  CONT_WR=67328,  CONT_BL=83712;
static const size_t MIDE_WL=83840,  MIDE_WR=100224, MIDE_BL=116608;
static const size_t UBIC_WL=116736, UBIC_WR=133120, UBIC_BL=149504;

static inline unsigned nb(long long n) { return (unsigned)((n + 255) / 256); }

static inline void gemm(hipStream_t st, const float* A, int lda, const float* W,
                        const float* bias, float* C, int N, int K, int NOa,
                        int NOp, int flags) {
    unsigned blocks = (unsigned)((N + 255) / 256);
    if (NOp == 128)     gemm_wmma<8><<<blocks, 256, 0, st>>>(A, lda, W, bias, C, N, K, NOa, flags);
    else if (NOp == 64) gemm_wmma<4><<<blocks, 256, 0, st>>>(A, lda, W, bias, C, N, K, NOa, flags);
    else                gemm_wmma<1><<<blocks, 256, 0, st>>>(A, lda, W, bias, C, N, K, NOa, flags);
}

extern "C" void kernel_launch(void* const* d_in, const int* in_sizes, int n_in,
                              void* d_out, int out_size, void* d_ws, size_t ws_size,
                              hipStream_t stream)
{
    (void)n_in; (void)out_size; (void)ws_size;
    const float* xc = (const float*)d_in[0];   // contador        [Nc,128]
    const float* xs = (const float*)d_in[1];   // suministro      [Ns, 64]
    const float* xm = (const float*)d_in[2];   // comercializadora[Nm, 32]
    const float* xu = (const float*)d_in[3];   // ubicacion       [Nu, 16]
    const float* xk = (const float*)d_in[4];   // concentrador    [Nk, 32]
    const int* eM = (const int*)d_in[6];  int EM = in_sizes[6] / 2;  // mide
    const int* eU = (const int*)d_in[7];  int EU = in_sizes[7] / 2;  // ubicado
    const int* eK = (const int*)d_in[8];  int EK = in_sizes[8] / 2;  // comunica
    const int* eC = (const int*)d_in[9];  int EC = in_sizes[9] / 2;  // contratado
    const int* eG = (const int*)d_in[10]; int EG = in_sizes[10] / 2; // cerca
    const float* P = (const float*)d_in[11];

    const int Nc = in_sizes[0] / 128, Ns = in_sizes[1] / 64, Nm = in_sizes[2] / 32;
    const int Nu = in_sizes[3] / 16,  Nk = in_sizes[4] / 32;

    // --- workspace carve-out (floats) ---
    float* wsp = (float*)d_ws; size_t o = 0;
    auto carve = [&](size_t n) { float* p = wsp + o; o += n; return p; };
    float* Hc[2] = { carve((size_t)Nc * 128), carve((size_t)Nc * 128) };
    float* Hs[2] = { carve((size_t)Ns * 128), carve((size_t)Ns * 128) };
    float* Hu[2] = { carve((size_t)Nu * 128), carve((size_t)Nu * 128) };
    float* Hk[2] = { carve((size_t)Nk * 128), carve((size_t)Nk * 128) };
    float* Hm[2] = { carve((size_t)Nm * 128), carve((size_t)Nm * 128) };
    float* AGG  = carve((size_t)Nc * 128);   // agg / GAT hs / cls hidden-1
    float* CNT  = carve((size_t)Nc);
    float* ASRC = carve((size_t)Nc * 4);
    float* ADST = carve((size_t)Nc * 4);
    float* MB   = carve((size_t)Nc * 4);     // flipped-uint max, then f32 m
    float* DEN  = carve((size_t)Nc * 4);
    float* EXB  = carve((size_t)EG * 4);
    float* CTXW = carve((size_t)Nc * 128);
    float* CLS2 = carve((size_t)Nc * 64);
    float* LOGI = carve((size_t)Nc * 3);
    float* WSM  = carve(8);

    auto zf = [&](float* p, long long n) { fill_f32<<<nb(n), 256, 0, stream>>>(p, 0.0f, n); };

    // ---------------- embeddings (ReLU(x @ W + b)) ----------------
    gemm(stream, xc, 128, P + EMB_CTA_W, P + EMB_CTA_B, Hc[0], Nc, 128, 128, 128, GF_BIAS | GF_RELU);
    gemm(stream, xs,  64, P + EMB_SUM_W, P + EMB_SUM_B, Hs[0], Ns,  64, 128, 128, GF_BIAS | GF_RELU);
    gemm(stream, xu,  16, P + EMB_UBI_W, P + EMB_UBI_B, Hu[0], Nu,  16, 128, 128, GF_BIAS | GF_RELU);
    gemm(stream, xk,  32, P + EMB_CON_W, P + EMB_CON_B, Hk[0], Nk,  32, 128, 128, GF_BIAS | GF_RELU);
    gemm(stream, xm,  32, P + EMB_COM_W, P + EMB_COM_B, Hm[0], Nm,  32, 128, 128, GF_BIAS | GF_RELU);
    // (expediente_fraude embedding is dead code in the reference: skipped)

    // ---------------- 4 hetero layers ----------------
    int cur = 0;
    for (int l = 0; l < 4; ++l) {
        const size_t LB = LBASE + (size_t)l * LSTRIDE;
        struct SD { const int* ei; int E; float* hsrc; float* hdst; float* hout; int Nd;
                    size_t wl, wr, bl, bg, bb; };
        SD sd[4] = {
            { eM, EM, Hc[cur], Hs[cur], Hs[1-cur], Ns, LB+MIDE_WL, LB+MIDE_WR, LB+MIDE_BL, LB+BN_SUM_G, LB+BN_SUM_B },
            { eU, EU, Hc[cur], Hu[cur], Hu[1-cur], Nu, LB+UBIC_WL, LB+UBIC_WR, LB+UBIC_BL, LB+BN_UBI_G, LB+BN_UBI_B },
            { eK, EK, Hc[cur], Hk[cur], Hk[1-cur], Nk, LB+COMU_WL, LB+COMU_WR, LB+COMU_BL, LB+BN_CON_G, LB+BN_CON_B },
            { eC, EC, Hs[cur], Hm[cur], Hm[1-cur], Nm, LB+CONT_WL, LB+CONT_WR, LB+CONT_BL, LB+BN_COM_G, LB+BN_COM_B },
        };
        for (int t = 0; t < 4; ++t) {
            SD& d = sd[t];
            zf(AGG, (long long)d.Nd * 128); zf(CNT, d.Nd);
            seg_count<<<nb(d.E), 256, 0, stream>>>(d.ei + d.E, d.E, CNT);
            seg_scatter_sum<<<nb((long long)d.E * 128), 256, 0, stream>>>(d.hsrc, d.ei, d.ei + d.E, d.E, AGG);
            seg_normalize<<<nb((long long)d.Nd * 128), 256, 0, stream>>>(AGG, CNT, (long long)d.Nd * 128);
            gemm(stream, AGG,   128, P + d.wl, P + d.bl, d.hout, d.Nd, 128, 128, 128, GF_BIAS);
            gemm(stream, d.hdst,128, P + d.wr, nullptr,  d.hout, d.Nd, 128, 128, 128, GF_ACC);
            bn_relu_k<<<nb((long long)d.Nd * 128), 256, 0, stream>>>(d.hout, P + d.bg, P + d.bb,
                                                                     (long long)d.Nd * 128);
        }
        // GAT on 'cerca' (contador -> contador)
        float* HS = AGG;
        gemm(stream, Hc[cur], 128, P + LB + GAT_W, nullptr, HS, Nc, 128, 128, 128, 0);
        gat_scores_k<<<nb((long long)Nc * 4), 256, 0, stream>>>(HS, P + LB + GAT_ATT_S,
                                                                P + LB + GAT_ATT_D, ASRC, ADST, Nc);
        fill_u32<<<nb((long long)Nc * 4), 256, 0, stream>>>((unsigned*)MB, 0x007FFFFFu, (long long)Nc * 4);
        gat_edge_max_k<<<nb((long long)EG * 4), 256, 0, stream>>>(ASRC, ADST, eG, EG, (unsigned*)MB);
        gat_m_fin_k<<<nb((long long)Nc * 4), 256, 0, stream>>>((unsigned*)MB, MB, DEN, (long long)Nc * 4);
        gat_edge_exp_k<<<nb((long long)EG * 4), 256, 0, stream>>>(ASRC, ADST, eG, EG, MB, DEN, EXB);
        zf(Hc[1 - cur], (long long)Nc * 128);
        gat_scatter_k<<<nb((long long)EG * 128), 256, 0, stream>>>(HS, eG, EG, EXB, DEN, Hc[1 - cur]);
        gat_bn_relu_k<<<nb((long long)Nc * 128), 256, 0, stream>>>(Hc[1 - cur], P + LB + GAT_B,
                                                                   P + LB + BN_CTA_G, P + LB + BN_CTA_B,
                                                                   (long long)Nc * 128);
        cur = 1 - cur;
    } // cur == 0 after 4 layers

    // ---------------- contextual aggregation (reversed edges, weighted by softmax(info_att)) ----
    softmax5_k<<<1, 32, 0, stream>>>(P + INFO_ATT, WSM);
    zf(CTXW, (long long)Nc * 128);
    struct CD { const int* ei; int E; const float* feat; int slot; };
    CD cd[4] = { { eM, EM, Hs[cur], 0 }, { eU, EU, Hu[cur], 1 },
                 { eK, EK, Hk[cur], 2 }, { eG, EG, Hc[cur], 4 } };
    for (int t = 0; t < 4; ++t) {
        CD& c = cd[t];
        zf(AGG, (long long)Nc * 128); zf(CNT, Nc);
        seg_count<<<nb(c.E), 256, 0, stream>>>(c.ei, c.E, CNT);                       // scatter to src (row0)
        seg_scatter_sum<<<nb((long long)c.E * 128), 256, 0, stream>>>(c.feat, c.ei + c.E, c.ei, c.E, AGG);
        ctx_axpy_k<<<nb((long long)Nc * 128), 256, 0, stream>>>(CTXW, AGG, CNT, WSM, c.slot,
                                                                (long long)Nc * 128);
    }

    // ---------------- classifier: z = [hc | ctx] @ W1 .. log_softmax ----------------
    float* Z1 = AGG;
    gemm(stream, Hc[cur], 128, P + CLS_W1,             P + CLS_B1, Z1, Nc, 128, 128, 128, GF_BIAS);
    gemm(stream, CTXW,    128, P + CLS_W1 + 128 * 128, nullptr,    Z1, Nc, 128, 128, 128, GF_ACC | GF_RELU);
    gemm(stream, Z1,      128, P + CLS_W2, P + CLS_B2, CLS2, Nc, 128, 64, 64, GF_BIAS | GF_RELU);
    gemm(stream, CLS2,     64, P + CLS_W3, P + CLS_B3, LOGI, Nc,  64,  3, 16, GF_BIAS);
    logsoftmax3_k<<<nb(Nc), 256, 0, stream>>>(LOGI, (float*)d_out, Nc);
}
